// OneBitLinear_15805479649501
// MI455X (gfx1250) — matrix operand order repaired and checked by reading, not compiled, not run on hardware
//
#include <hip/hip_runtime.h>
#include <stdint.h>

// ---------------------------------------------------------------------------
// OneBitLinear forward for MI455X (gfx1250, wave32, WMMA).
//   out[M,N] = x[M,K] * w_eff[N,K]^T,  w_eff = sign(w)*scale(per 128-group)
// M=8192 N=11008 K=4096. Compute-bound (AI ~1100 flop/B) -> WMMA f16/f32-acc.
// Pre-converted f16 operands (157 MB) are L2-resident (192 MB L2).
// ---------------------------------------------------------------------------

typedef __attribute__((ext_vector_type(16))) _Float16     v16h;
typedef __attribute__((ext_vector_type(8)))  float        v8f;
typedef __attribute__((ext_vector_type(4)))  _Float16     h4;
typedef __attribute__((ext_vector_type(4)))  unsigned int u32x4;
typedef __attribute__((ext_vector_type(4)))  int          i32x4;

#define AS1 __attribute__((address_space(1)))
#define AS3 __attribute__((address_space(3)))

#define GSZ   128      // quantization group size
#define K_DIM 4096     // inner dim (fixed by problem)
#define BM    128      // block tile M
#define BN    128      // block tile N
#define BK    32       // k-step (matches WMMA K for f16)
#define LDT   40       // padded LDS row stride in f16 elems (80B = 5*16B, conflict-free)
#define NKSTEP (K_DIM / BK)

// ---- CDNA5 async global->LDS path (guarded; sync fallback always compiles) ----
#if defined(__gfx1250__) && __has_builtin(__builtin_amdgcn_global_load_async_to_lds_b128)
#define HAVE_ASYNC_LDS 1
#else
#define HAVE_ASYNC_LDS 0
#endif

#if defined(__gfx1250__) && __has_builtin(__builtin_amdgcn_s_wait_asynccnt)
#define WAIT_ASYNC(n) __builtin_amdgcn_s_wait_asynccnt(n)
#elif HAVE_ASYNC_LDS
#define WAIT_ASYNC(n) asm volatile("s_wait_asynccnt %0" ::"n"(n))
#else
#define WAIT_ASYNC(n) do {} while (0)
#endif

// ---------------------------------------------------------------------------
// Kernel 1: per-128-group L-inf scale + sign binarization -> f16 w_eff[N][K].
// One wave32 per group: 32 lanes x float4 = 128 elems.
// ---------------------------------------------------------------------------
__global__ __launch_bounds__(256)
void binarize_weight(const float* __restrict__ w, _Float16* __restrict__ weff,
                     int total_groups) {
  const int gidx = (int)((blockIdx.x * blockDim.x + threadIdx.x) >> 5);
  const int lane = threadIdx.x & 31;
  if (gidx >= total_groups) return;

  const float4 v = ((const float4*)(w + (size_t)gidx * GSZ))[lane];
  float m = fmaxf(fmaxf(fabsf(v.x), fabsf(v.y)), fmaxf(fabsf(v.z), fabsf(v.w)));
#pragma unroll
  for (int off = 16; off > 0; off >>= 1)
    m = fmaxf(m, __shfl_xor(m, off, 32));
  const float s = fmaxf(m, 1e-8f);

  h4 r;
  r.x = (_Float16)(v.x > 0.f ? s : (v.x < 0.f ? -s : 0.f));
  r.y = (_Float16)(v.y > 0.f ? s : (v.y < 0.f ? -s : 0.f));
  r.z = (_Float16)(v.z > 0.f ? s : (v.z < 0.f ? -s : 0.f));
  r.w = (_Float16)(v.w > 0.f ? s : (v.w < 0.f ? -s : 0.f));
  ((h4*)(weff + (size_t)gidx * GSZ))[lane] = r;
}

// ---------------------------------------------------------------------------
// Kernel 2: x fp32 -> fp16 (activations ~N(0,1) fit f16 easily).
// ---------------------------------------------------------------------------
__global__ __launch_bounds__(256)
void convert_x(const float* __restrict__ x, _Float16* __restrict__ xh, size_t n4) {
  const size_t i = (size_t)blockIdx.x * blockDim.x + threadIdx.x;
  if (i >= n4) return;
  const float4 v = ((const float4*)x)[i];
  h4 r;
  r.x = (_Float16)v.x; r.y = (_Float16)v.y;
  r.z = (_Float16)v.z; r.w = (_Float16)v.w;
  ((h4*)xh)[i] = r;
}

// ---------------------------------------------------------------------------
// Kernel 3: WMMA GEMM. A = xh[M][K], B = weff[N][K] (both row-major over K,
// which is exactly the WMMA-native layout). 256 threads = 8 waves in a 2x4
// grid; wave tile 64x32 = 4x2 tiles of v_wmma_f32_16x16x32_f16.
// Double-buffered LDS staging via async global->LDS (CDNA5 async DMA path).
// ---------------------------------------------------------------------------
__global__ __launch_bounds__(256)
void onebit_gemm(const _Float16* __restrict__ A, const _Float16* __restrict__ B,
                 float* __restrict__ C, int M, int N) {
  __shared__ _Float16 As[2][BM * LDT];   // 20480 B
  __shared__ _Float16 Bs[2][BN * LDT];   // 20480 B

  const int tid  = threadIdx.x;
  const int lane = tid & 31;
  const int wave = tid >> 5;
  const int wm   = wave >> 2;      // 0..1 : wave row (64 M-rows each)
  const int wn   = wave & 3;       // 0..3 : wave col (32 N-cols each)
  const int half = lane >> 4;      // 0/1 : half-wave (K-group select)
  const int lrow = lane & 15;      // row/col within 16x16 tile

  const int blockN = blockIdx.x * BN;
  const int blockM = blockIdx.y * BM;

  // Staging: tile is 128 rows x 32 f16 = 128 x four 16B chunks.
  // Thread t moves chunks {t, t+256} of A and of B (4 x b128 per thread).
  const int row0 = tid >> 2;            // 0..63
  const int cc   = tid & 3;             // 16B chunk within row

  const _Float16* gA0 = A + (size_t)(blockM + row0) * K_DIM + cc * 8;
  const _Float16* gA1 = gA0 + (size_t)64 * K_DIM;
  const _Float16* gB0 = B + (size_t)(blockN + row0) * K_DIM + cc * 8;
  const _Float16* gB1 = gB0 + (size_t)64 * K_DIM;

  const int lOffA0 = row0 * LDT + cc * 8;
  const int lOffA1 = (row0 + 64) * LDT + cc * 8;

  auto stage = [&](int buf, int kt) {
    const int ko = kt * BK;
#if HAVE_ASYNC_LDS
    __builtin_amdgcn_global_load_async_to_lds_b128(
        (AS1 i32x4*)(gA0 + ko), (AS3 i32x4*)(&As[buf][lOffA0]), 0, 0);
    __builtin_amdgcn_global_load_async_to_lds_b128(
        (AS1 i32x4*)(gA1 + ko), (AS3 i32x4*)(&As[buf][lOffA1]), 0, 0);
    __builtin_amdgcn_global_load_async_to_lds_b128(
        (AS1 i32x4*)(gB0 + ko), (AS3 i32x4*)(&Bs[buf][lOffA0]), 0, 0);
    __builtin_amdgcn_global_load_async_to_lds_b128(
        (AS1 i32x4*)(gB1 + ko), (AS3 i32x4*)(&Bs[buf][lOffA1]), 0, 0);
#else
    const u32x4 a0 = *(const u32x4*)(gA0 + ko);
    const u32x4 a1 = *(const u32x4*)(gA1 + ko);
    const u32x4 b0 = *(const u32x4*)(gB0 + ko);
    const u32x4 b1 = *(const u32x4*)(gB1 + ko);
    *(u32x4*)(&As[buf][lOffA0]) = a0;
    *(u32x4*)(&As[buf][lOffA1]) = a1;
    *(u32x4*)(&Bs[buf][lOffA0]) = b0;
    *(u32x4*)(&Bs[buf][lOffA1]) = b1;
#endif
  };

  union Frag { v16h v; u32x4 u[2]; };
  v8f acc[4][2] = {};

  auto compute = [&](int buf) {
    const _Float16* as = As[buf];
    const _Float16* bs = Bs[buf];
    Frag a[4], b[2];
    // A frag (16x32 f16, ISA layout): lanes 0-15 hold K 0-7 & 16-23,
    // lanes 16-31 hold K 8-15 & 24-31, row = lane%16.
    const int klo = half * 8;
#pragma unroll
    for (int i = 0; i < 4; ++i) {
      const _Float16* p = as + (wm * 64 + i * 16 + lrow) * LDT;
      a[i].u[0] = *(const u32x4*)(p + klo);
      a[i].u[1] = *(const u32x4*)(p + klo + 16);
    }
    // B frag (32x16 f16): lanes 0-15 hold K 0-7 & 16-23, lanes 16-31 hold K 8-15 & 24-31,
    // col = lane%16; B stored as [N][K].
#pragma unroll
    for (int j = 0; j < 2; ++j) {
      const _Float16* p = bs + (wn * 32 + j * 16 + lrow) * LDT + half * 8;
      b[j].u[0] = *(const u32x4*)(p);
      b[j].u[1] = *(const u32x4*)(p + 16);
    }
#pragma unroll
    for (int i = 0; i < 4; ++i)
#pragma unroll
      for (int j = 0; j < 2; ++j)
        acc[i][j] = __builtin_amdgcn_wmma_f32_16x16x32_f16(
            false, a[i].v, false, b[j].v, (short)0, acc[i][j], false, false);
  };

  stage(0, 0);
#pragma unroll 1
  for (int kt = 0; kt < NKSTEP; ++kt) {
    const int cur = kt & 1;
    if (kt + 1 < NKSTEP) {
      stage(cur ^ 1, kt + 1);  // overlap next-stage DMA with compute
      WAIT_ASYNC(4);           // the 4 just-issued may remain outstanding
    } else {
      WAIT_ASYNC(0);
    }
    __syncthreads();
    compute(cur);
    __syncthreads();
  }

  // Epilogue: C/D layout -> VGPR r holds (M = r + 8*half), N = lane%16.
#pragma unroll
  for (int i = 0; i < 4; ++i)
#pragma unroll
    for (int j = 0; j < 2; ++j) {
      const int mg = blockM + wm * 64 + i * 16 + half * 8;
      const int ng = blockN + wn * 32 + j * 16 + lrow;
      float* cp = C + (size_t)mg * N + ng;
#pragma unroll
      for (int r = 0; r < 8; ++r)
        cp[(size_t)r * N] = acc[i][j][r];
    }
}

// ---------------------------------------------------------------------------
extern "C" void kernel_launch(void* const* d_in, const int* in_sizes, int n_in,
                              void* d_out, int out_size, void* d_ws, size_t ws_size,
                              hipStream_t stream) {
  const float* x = (const float*)d_in[0];   // [4,2048,4096] fp32
  const float* w = (const float*)d_in[1];   // [11008,4096] fp32
  float* out = (float*)d_out;               // [4,2048,11008] fp32

  const int K = K_DIM;
  const int Mtot = in_sizes[0] / K;         // 8192
  const int O    = in_sizes[1] / K;         // 11008

  // Workspace: w_eff f16 (90.2 MB) then x f16 (67.1 MB) -> 157 MB total.
  _Float16* weff = (_Float16*)d_ws;
  _Float16* xh   = (_Float16*)((char*)d_ws + (size_t)O * K * sizeof(_Float16));

  const int total_groups = (int)(((size_t)O * K) / GSZ);          // 352256
  const int bin_blocks   = (total_groups * 32 + 255) / 256;       // 44032
  binarize_weight<<<bin_blocks, 256, 0, stream>>>(w, weff, total_groups);

  const size_t n4 = (size_t)Mtot * K / 4;                         // 8388608
  convert_x<<<(unsigned)((n4 + 255) / 256), 256, 0, stream>>>(x, xh, n4);

  dim3 grid(O / BN, Mtot / BM);                                   // (86, 64)
  onebit_gemm<<<grid, 256, 0, stream>>>(xh, weff, out, Mtot, O);
}
